// DynamicConvolutionalBlockedAttention_52639119180094
// MI455X (gfx1250) — compile-verified, matrix operand order checked by reading, not run on hardware
//
#include <hip/hip_runtime.h>
#include <hip/hip_bf16.h>
#include <cstdint>

typedef __bf16 bf16_t;
typedef __attribute__((ext_vector_type(16))) __bf16 v16bf;
typedef __attribute__((ext_vector_type(8)))  __bf16 v8bf;
typedef __attribute__((ext_vector_type(8)))  float  v8f;

#define B_   8
#define H_   8
#define L_   1024
#define C_   1024
#define DH_  128

#define BM 128
#define BN 128
#define BK 32
#define NTHREADS 256
#define CHUNK 8   // (b,h) heads per score-buffer pass (8 passes total)

// ---------------------------------------------------------------------------
// Fragment loader for v_wmma_f32_16x16x32_bf16, TN layout.
// 16-bit A-matrix 16x32 layout (ISA 7.12.2): lane L<16 holds row M=L with
// K = {kb..kb+7, kb+16..kb+23}, kb = 0; lanes 16..31 same rows with kb = 8.
// B (32x16 KxN) mirrors this when the operand is stored row-major NxK
// (i.e. we always feed B^T rows), so one loader serves both operands.
// ---------------------------------------------------------------------------
__device__ __forceinline__ v16bf load_frag16(const bf16_t* base, int ld) {
  const int lane = threadIdx.x & 31;
  const int row  = lane & 15;
  const int kb   = (lane & 16) >> 1;          // 0 or 8
  const bf16_t* p = base + (size_t)row * ld + kb;
  v8bf lo = *reinterpret_cast<const v8bf*>(p);        // K = kb .. kb+7
  v8bf hi = *reinterpret_cast<const v8bf*>(p + 16);   // K = kb+16 .. kb+23
  v16bf f;
#pragma unroll
  for (int i = 0; i < 8; ++i) { f[i] = lo[i]; f[i + 8] = hi[i]; }
  return f;
}

// ---------------------------------------------------------------------------
// Core 128x128 block (8 waves: 4x2 grid of 32x64 wave tiles), K step 32.
// B-operand slab (128 rows x 32 K, 8 KB bf16) is staged into LDS with the
// gfx1250 async DMA (global_load_async_to_lds_b128, ASYNCcnt) and consumed
// from LDS; A fragments stream directly from global (L2-resident, x2 reuse).
// ---------------------------------------------------------------------------
__device__ __forceinline__ void wmma_block(v8f acc[2][4],
    const bf16_t* __restrict__ A, int lda,
    const bf16_t* B, int ldb,
    int aRow0, int bn0, int bRowLocal, int kBeg, int kEnd) {
  __shared__ bf16_t ldsB[BN * BK];            // 8 KB
  const int tid  = threadIdx.x;
  const int lane = tid & 31;

  for (int k = kBeg; k < kEnd; k += BK) {
    // --- async stage of B rows bn0..bn0+127, cols k..k+31 (512 x 16B chunks)
#pragma unroll
    for (int u = 0; u < 2; ++u) {
      const int c = tid * 2 + u;              // 0..511
      const int r = c >> 2;                   // local row 0..127
      const int q = c & 3;                    // 16-byte quarter of the row
      const bf16_t* src = B + (size_t)(bn0 + r) * ldb + k + q * 8;
      const uint32_t dst = (uint32_t)(uintptr_t)&ldsB[r * BK + q * 8];
      asm volatile("global_load_async_to_lds_b128 %0, %1, off"
                   :: "v"(dst), "v"(src) : "memory");
    }
    if (k + BK < kEnd) {  // near-scope prefetch of the next A slab
      __builtin_prefetch(A + (size_t)(aRow0 + lane) * lda + k + BK, 0, 3);
    }
    asm volatile("s_wait_asynccnt 0" ::: "memory");
    __syncthreads();                          // LDS slab visible to all waves

    v16bf af[2], bfv[4];
#pragma unroll
    for (int i = 0; i < 2; ++i)
      af[i] = load_frag16(A + (size_t)(aRow0 + 16 * i) * lda + k, lda);
#pragma unroll
    for (int j = 0; j < 4; ++j)
      bfv[j] = load_frag16(ldsB + (bRowLocal + 16 * j) * BK, BK);
#pragma unroll
    for (int i = 0; i < 2; ++i)
#pragma unroll
      for (int j = 0; j < 4; ++j)
        acc[i][j] = __builtin_amdgcn_wmma_f32_16x16x32_bf16(
            false, af[i], false, bfv[j], (short)0, acc[i][j], false, false);
    __syncthreads();                          // protect slab before overwrite
  }
}

// C/D 16x16 f32 layout: lane<16 -> (M=e, N=lane); lane>=16 -> (M=8+e, N=lane-16)
template <typename F>
__device__ __forceinline__ void epilogue_for_each(const v8f acc[2][4],
                                                  int aRow0, int bRow0, F&& f) {
  const int lane = threadIdx.x & 31;
  const int nl = lane & 15, mh = (lane >> 4) * 8;
#pragma unroll
  for (int i = 0; i < 2; ++i)
#pragma unroll
    for (int j = 0; j < 4; ++j) {
      const int tM = aRow0 + i * 16, tN = bRow0 + j * 16;
#pragma unroll
      for (int e = 0; e < 8; ++e)
        f(tM + mh + e, tN + nl, acc[i][j][e]);
    }
}

// ---------------------------------------------------------------------------
// fp32 -> bf16 convert
// ---------------------------------------------------------------------------
__global__ void cvt_kernel(bf16_t* __restrict__ dst, const float* __restrict__ src, int n) {
  int i = blockIdx.x * blockDim.x + threadIdx.x;
  const int stride = gridDim.x * blockDim.x;
  for (; i < n; i += stride) dst[i] = (bf16_t)src[i];
}

// ---------------------------------------------------------------------------
// QKV projection: out(M=8192,N=1024) = A(8192x1024) . W^T(row-major W) + bias
// vtmode=1 stores V transposed per head: vt[b][h*128+d][l] = V[b][l][h*128+d]
// ---------------------------------------------------------------------------
__global__ __launch_bounds__(NTHREADS)
void gemm_qkv_kernel(bf16_t* __restrict__ out, const bf16_t* __restrict__ A,
                     const bf16_t* __restrict__ W, const float* __restrict__ bias,
                     int vtmode) {
  const int tid = threadIdx.x;
  const int wid = tid >> 5, wm = wid & 3, wn = wid >> 2;
  const int bm0 = blockIdx.x * BM, bn0 = blockIdx.y * BN;
  const int aRow0 = bm0 + wm * 32, bRow0 = bn0 + wn * 64;
  v8f z = {}; v8f acc[2][4];
#pragma unroll
  for (int i = 0; i < 2; ++i)
#pragma unroll
    for (int j = 0; j < 4; ++j) acc[i][j] = z;
  wmma_block(acc, A, C_, W, C_, aRow0, bn0, wn * 64, 0, C_);
  epilogue_for_each(acc, aRow0, bRow0, [&](int m, int n, float accv) {
    const float val = accv + bias[n];
    if (!vtmode) {
      out[(size_t)m * C_ + n] = (bf16_t)val;
    } else {
      const int bb = m >> 10;
      const int l  = m & (L_ - 1);
      out[((size_t)(bb * C_ + n)) * L_ + l] = (bf16_t)val;
    }
  });
}

// ---------------------------------------------------------------------------
// Scores: S = (q . k^T) * scale, zero-filled above the diagonal (for conv).
// Tiles fully above the diagonal skip the K loop entirely.
// ---------------------------------------------------------------------------
__global__ __launch_bounds__(NTHREADS)
void qk_scores_kernel(float* __restrict__ Sc, const bf16_t* __restrict__ Q,
                      const bf16_t* __restrict__ Kb, int g0, float scale) {
  const int gg = blockIdx.z, g = g0 + gg, bh = g >> 3, hh = g & 7;
  const int tid = threadIdx.x;
  const int wid = tid >> 5, wm = wid & 3, wn = wid >> 2;
  const int bm0 = blockIdx.x * BM, bn0 = blockIdx.y * BN;
  const int aRow0 = bm0 + wm * 32, bRow0 = bn0 + wn * 64;
  float* __restrict__ out = Sc + (size_t)gg * L_ * L_;

  if (bn0 > bm0 + (BM - 1)) {  // entirely above diagonal -> store zeros
    const int lane = tid & 31, nl = lane & 15, mh = (lane >> 4) * 8;
#pragma unroll
    for (int i = 0; i < 2; ++i)
#pragma unroll
      for (int j = 0; j < 4; ++j) {
        const int tM = aRow0 + i * 16, tN = bRow0 + j * 16;
#pragma unroll
        for (int e = 0; e < 8; ++e)
          out[(size_t)(tM + mh + e) * L_ + tN + nl] = 0.0f;
      }
    return;
  }
  const bf16_t* Aq = Q  + (size_t)bh * L_ * C_ + hh * DH_;
  const bf16_t* Ak = Kb + (size_t)bh * L_ * C_ + hh * DH_;
  v8f z = {}; v8f acc[2][4];
#pragma unroll
  for (int i = 0; i < 2; ++i)
#pragma unroll
    for (int j = 0; j < 4; ++j) acc[i][j] = z;
  wmma_block(acc, Aq, C_, Ak, C_, aRow0, bn0, wn * 64, 0, DH_);
  epilogue_for_each(acc, aRow0, bRow0, [&](int m, int n, float accv) {
    float val = accv * scale;
    if (n > m) val = 0.0f;           // causal zero-fill (conv input)
    out[(size_t)m * L_ + n] = val;
  });
}

// ---------------------------------------------------------------------------
// Causal-masked depthwise conv (5 surviving taps: needs rows i-1, i only)
// + re-mask (-1e9) + softmax over j, emit probs in bf16.
// Conv channel index = b (reference convolves (H,B,L,L) with groups=B).
// One block per (head-in-chunk, row i); 256 threads x 4 columns.
// ---------------------------------------------------------------------------
__global__ __launch_bounds__(NTHREADS)
void conv_softmax_kernel(bf16_t* __restrict__ probs, const float* __restrict__ scores,
                         const float* __restrict__ conv_w, const float* __restrict__ conv_b,
                         int g0) {
  const int i  = blockIdx.x;
  const int gg = blockIdx.y;
  const int g  = g0 + gg;
  const int b  = g >> 3;                 // conv channel
  const int tid = threadIdx.x;
  const float w00 = conv_w[b * 9 + 0], w01 = conv_w[b * 9 + 1], w02 = conv_w[b * 9 + 2];
  const float w10 = conv_w[b * 9 + 3], w11 = conv_w[b * 9 + 4];
  const float cb  = conv_b[b];

  const float* __restrict__ cur = scores + ((size_t)gg * L_ + i) * L_;
  const float* __restrict__ prv = cur - L_;
  const int j0 = tid * 4;

  const float c0 = cur[j0 + 0], c1 = cur[j0 + 1], c2 = cur[j0 + 2], c3 = cur[j0 + 3];
  const float cl = (j0 > 0) ? cur[j0 - 1] : 0.0f;
  float p0 = 0.f, p1 = 0.f, p2 = 0.f, p3 = 0.f, pl = 0.f, pr = 0.f;
  if (i > 0) {
    p0 = prv[j0 + 0]; p1 = prv[j0 + 1]; p2 = prv[j0 + 2]; p3 = prv[j0 + 3];
    pl = (j0 > 0) ? prv[j0 - 1] : 0.0f;
    pr = (j0 + 4 < L_) ? prv[j0 + 4] : 0.0f;
  }
  float v[4];
  v[0] = w00 * pl + w01 * p0 + w02 * p1 + w10 * cl + w11 * c0 + cb;
  v[1] = w00 * p0 + w01 * p1 + w02 * p2 + w10 * c0 + w11 * c1 + cb;
  v[2] = w00 * p1 + w01 * p2 + w02 * p3 + w10 * c1 + w11 * c2 + cb;
  v[3] = w00 * p2 + w01 * p3 + w02 * pr + w10 * c2 + w11 * c3 + cb;
#pragma unroll
  for (int t = 0; t < 4; ++t)
    if (j0 + t > i) v[t] = -1e9f;        // causal re-mask before softmax

  __shared__ float red[NTHREADS];
  float lmax = fmaxf(fmaxf(v[0], v[1]), fmaxf(v[2], v[3]));
  red[tid] = lmax; __syncthreads();
  for (int s = NTHREADS / 2; s > 0; s >>= 1) {
    if (tid < s) red[tid] = fmaxf(red[tid], red[tid + s]);
    __syncthreads();
  }
  const float mrow = red[0]; __syncthreads();

  float e[4]; float lsum = 0.0f;
#pragma unroll
  for (int t = 0; t < 4; ++t) { e[t] = __expf(v[t] - mrow); lsum += e[t]; }
  red[tid] = lsum; __syncthreads();
  for (int s = NTHREADS / 2; s > 0; s >>= 1) {
    if (tid < s) red[tid] += red[tid + s];
    __syncthreads();
  }
  const float inv = 1.0f / red[0];

  bf16_t* __restrict__ orow = probs + ((size_t)gg * L_ + i) * L_ + j0;
#pragma unroll
  for (int t = 0; t < 4; ++t) orow[t] = (bf16_t)(e[t] * inv);
}

// ---------------------------------------------------------------------------
// P.V: out(1024x128) = probs(1024x1024) . V; B operand is V^T (128x1024,
// K-contiguous). Causal probs -> K loop truncated at row-block end.
// ---------------------------------------------------------------------------
__global__ __launch_bounds__(NTHREADS)
void pv_kernel(bf16_t* __restrict__ attn, const bf16_t* __restrict__ P,
               const bf16_t* __restrict__ Vt, int g0) {
  const int gg = blockIdx.z, g = g0 + gg, bh = g >> 3, hh = g & 7;
  const int tid = threadIdx.x;
  const int wid = tid >> 5, wm = wid & 3, wn = wid >> 2;
  const int bm0 = blockIdx.x * BM;
  const int aRow0 = bm0 + wm * 32, bRow0 = wn * 64;
  const bf16_t* A  = P  + (size_t)gg * L_ * L_;
  const bf16_t* Bv = Vt + (size_t)(bh * H_ + hh) * DH_ * L_;
  v8f z = {}; v8f acc[2][4];
#pragma unroll
  for (int i = 0; i < 2; ++i)
#pragma unroll
    for (int j = 0; j < 4; ++j) acc[i][j] = z;
  wmma_block(acc, A, L_, Bv, L_, aRow0, 0, wn * 64, 0, bm0 + BM);
  epilogue_for_each(acc, aRow0, bRow0, [&](int m, int n, float accv) {
    attn[((size_t)bh * L_ + m) * C_ + hh * DH_ + n] = (bf16_t)accv;
  });
}

// ---------------------------------------------------------------------------
// Output projection: d_out(8192x1024,f32) = attn . Wo^T + bo
// ---------------------------------------------------------------------------
__global__ __launch_bounds__(NTHREADS)
void gemm_out_kernel(float* __restrict__ out, const bf16_t* __restrict__ A,
                     const bf16_t* __restrict__ W, const float* __restrict__ bias) {
  const int tid = threadIdx.x;
  const int wid = tid >> 5, wm = wid & 3, wn = wid >> 2;
  const int bm0 = blockIdx.x * BM, bn0 = blockIdx.y * BN;
  const int aRow0 = bm0 + wm * 32, bRow0 = bn0 + wn * 64;
  v8f z = {}; v8f acc[2][4];
#pragma unroll
  for (int i = 0; i < 2; ++i)
#pragma unroll
    for (int j = 0; j < 4; ++j) acc[i][j] = z;
  wmma_block(acc, A, C_, W, C_, aRow0, bn0, wn * 64, 0, C_);
  epilogue_for_each(acc, aRow0, bRow0, [&](int m, int n, float accv) {
    out[(size_t)m * C_ + n] = accv + bias[n];
  });
}

// ---------------------------------------------------------------------------
extern "C" void kernel_launch(void* const* d_in, const int* in_sizes, int n_in,
                              void* d_out, int out_size, void* d_ws, size_t ws_size,
                              hipStream_t stream) {
  (void)in_sizes; (void)n_in; (void)out_size; (void)ws_size;
  const float* x  = (const float*)d_in[0];
  const float* Wq = (const float*)d_in[1];
  const float* bq = (const float*)d_in[2];
  const float* Wk = (const float*)d_in[3];
  const float* bk = (const float*)d_in[4];
  const float* Wv = (const float*)d_in[5];
  const float* bv = (const float*)d_in[6];
  const float* cw = (const float*)d_in[7];
  const float* cb = (const float*)d_in[8];
  const float* Wo = (const float*)d_in[9];
  const float* bo = (const float*)d_in[10];
  // d_in[11] = conv_times (== 1 in this setup; single conv application)

  size_t off = 0;
  char* base = (char*)d_ws;
  auto wsalloc = [&](size_t bytes) -> char* {
    char* p = base + off;
    off += (bytes + 255) & ~(size_t)255;
    return p;
  };
  bf16_t* xb      = (bf16_t*)wsalloc((size_t)B_ * L_ * C_ * 2);      // 16 MB
  bf16_t* wqb     = (bf16_t*)wsalloc((size_t)C_ * C_ * 2);           //  2 MB
  bf16_t* wkb     = (bf16_t*)wsalloc((size_t)C_ * C_ * 2);
  bf16_t* wvb     = (bf16_t*)wsalloc((size_t)C_ * C_ * 2);
  bf16_t* wob     = (bf16_t*)wsalloc((size_t)C_ * C_ * 2);
  bf16_t* qb      = (bf16_t*)wsalloc((size_t)B_ * L_ * C_ * 2);      // 16 MB
  bf16_t* kb      = (bf16_t*)wsalloc((size_t)B_ * L_ * C_ * 2);      // 16 MB
  bf16_t* vtb     = (bf16_t*)wsalloc((size_t)B_ * L_ * C_ * 2);      // 16 MB (B,H,dh,L)
  bf16_t* attnb   = (bf16_t*)wsalloc((size_t)B_ * L_ * C_ * 2);      // 16 MB
  float*  scoresf = (float*) wsalloc((size_t)CHUNK * L_ * L_ * 4);   // 32 MB
  bf16_t* probsb  = (bf16_t*)wsalloc((size_t)CHUNK * L_ * L_ * 2);   // 16 MB

  // fp32 -> bf16 conversions
  cvt_kernel<<<4096, 256, 0, stream>>>(xb,  x,  B_ * L_ * C_);
  cvt_kernel<<<1024, 256, 0, stream>>>(wqb, Wq, C_ * C_);
  cvt_kernel<<<1024, 256, 0, stream>>>(wkb, Wk, C_ * C_);
  cvt_kernel<<<1024, 256, 0, stream>>>(wvb, Wv, C_ * C_);
  cvt_kernel<<<1024, 256, 0, stream>>>(wob, Wo, C_ * C_);

  // QKV projections (V stored transposed per head)
  const dim3 gProj(B_ * L_ / BM, C_ / BN);
  gemm_qkv_kernel<<<gProj, NTHREADS, 0, stream>>>(qb,  xb, wqb, bq, 0);
  gemm_qkv_kernel<<<gProj, NTHREADS, 0, stream>>>(kb,  xb, wkb, bk, 0);
  gemm_qkv_kernel<<<gProj, NTHREADS, 0, stream>>>(vtb, xb, wvb, bv, 1);

  const float scale = 0.08838834764831845f;  // 1/sqrt(128)
  for (int c = 0; c < (B_ * H_) / CHUNK; ++c) {
    const int g0 = c * CHUNK;
    qk_scores_kernel<<<dim3(L_ / BM, L_ / BN, CHUNK), NTHREADS, 0, stream>>>(
        scoresf, qb, kb, g0, scale);
    conv_softmax_kernel<<<dim3(L_, CHUNK), NTHREADS, 0, stream>>>(
        probsb, scoresf, cw, cb, g0);
    pv_kernel<<<dim3(L_ / BM, 1, CHUNK), NTHREADS, 0, stream>>>(
        attnb, probsb, vtb, g0);
  }

  // Output projection (fp32 result)
  gemm_out_kernel<<<gProj, NTHREADS, 0, stream>>>((float*)d_out, attnb, wob, bo);
}
